// CayleyOrthogonalHyperConnection_58325655880244
// MI455X (gfx1250) — matrix operand order repaired and checked by reading, not compiled, not run on hardware
//
#include <hip/hip_runtime.h>
#include <hip/hip_bf16.h>

// Problem dims (fixed by the reference): b=2, l=4096, d=2048, n_streams=4
#define D_DIM 2048
#define M_DIM 8192      // b*l tokens
#define N_DIM 2048
#define K_DIM 2048
#define NG    48        // 3 * n * n gate outputs

// GEMM tiling: 256 threads = 8 wave32 waves; waves arranged 4(M) x 2(N)
#define BM 128
#define BN 128
#define BK 32
#define KTILES (K_DIM / BK)
#define LDB 56          // padded bf16 LDS row stride: 112B, multiple of 16B,
                        // (28*m) % 64 bank starts tile all 64 banks -> frag
                        // reads are bank-conflict-free

typedef __attribute__((ext_vector_type(16))) __bf16 v16bf;
typedef __attribute__((ext_vector_type(8)))  float  v8f;

union Frag { v16bf v; uint4 q[2]; };

__device__ __forceinline__ v8f wmma_bf16(v16bf a, v16bf b, v8f c) {
    // v_wmma_f32_16x16x32_bf16: D = A(16x32) * B(32x16) + C, fp32 accum
    return __builtin_amdgcn_wmma_f32_16x16x32_bf16(false, a, false, b,
                                                   (short)0, c, false, false);
}

__device__ __forceinline__ void split_bf16(float f, __bf16& h, __bf16& l) {
    h = (__bf16)f;
    l = (__bf16)(f - (float)h);
}

__device__ __forceinline__ unsigned pack2(__bf16 a, __bf16 b) {
    unsigned ua = (unsigned)__builtin_bit_cast(unsigned short, a);
    unsigned ub = (unsigned)__builtin_bit_cast(unsigned short, b);
    return ua | (ub << 16);
}

// ---------------------------------------------------------------------------
// Kernel 1: per-token coefficients.
// LN(x) -> gates(48) -> row-softmax / col-softmax / iterative Cayley ->
// c_res[m], c_mix[m] = c_pre[m]*c_post[m].
// Memory bound: reads x once (64 MB ~ 3us at 23.3 TB/s).
// ---------------------------------------------------------------------------
__global__ void __launch_bounds__(256)
coeff_kernel(const float* __restrict__ x, const float* __restrict__ ln_w,
             const float* __restrict__ ln_b, const float* __restrict__ proj_w,
             const float* __restrict__ proj_b,
             float* __restrict__ c_res, float* __restrict__ c_mix)
{
    const int tok = blockIdx.x;          // 0..8191
    const int tid = threadIdx.x;         // 256 threads = 8 waves
    const float* xr = x + (size_t)tok * D_DIM;

    __shared__ float xn[D_DIM];
    __shared__ float r1[256];
    __shared__ float r2[256];
    __shared__ float gates[NG];

    // 8 contiguous floats per thread (two 16B loads, lane-contiguous 32B)
    float4 a = *(const float4*)(xr + tid * 8);
    float4 b = *(const float4*)(xr + tid * 8 + 4);
    float s  = a.x + a.y + a.z + a.w + b.x + b.y + b.z + b.w;
    float ss = a.x*a.x + a.y*a.y + a.z*a.z + a.w*a.w
             + b.x*b.x + b.y*b.y + b.z*b.z + b.w*b.w;
    r1[tid] = s; r2[tid] = ss;
    __syncthreads();
    for (int st = 128; st > 0; st >>= 1) {
        if (tid < st) { r1[tid] += r1[tid + st]; r2[tid] += r2[tid + st]; }
        __syncthreads();
    }
    const float mu  = r1[0] * (1.0f / D_DIM);
    const float var = r2[0] * (1.0f / D_DIM) - mu * mu;
    const float inv = rsqrtf(var + 1e-5f);

    {
        const int base = tid * 8;
        const float v[8] = {a.x, a.y, a.z, a.w, b.x, b.y, b.z, b.w};
        #pragma unroll
        for (int i = 0; i < 8; ++i)
            xn[base + i] = (v[i] - mu) * inv * ln_w[base + i] + ln_b[base + i];
    }
    __syncthreads();

    // 48 gate dot products of length 2048 (proj_w is 384 KB -> L2 resident)
    if (tid < NG) {
        const float4* xv = (const float4*)xn;
        const float4* wv = (const float4*)(proj_w + (size_t)tid * D_DIM);
        float acc = proj_b[tid];
        for (int k = 0; k < D_DIM / 4; ++k) {
            float4 xa = xv[k], wa = wv[k];
            acc += xa.x * wa.x + xa.y * wa.y + xa.z * wa.z + xa.w * wa.w;
        }
        gates[tid] = acc;
    }
    __syncthreads();

    if (tid == 0) {
        // coeff_pre: mean_i sum_j softmax_row(pre)  (== 1 up to rounding)
        float cpre = 0.f;
        for (int i = 0; i < 4; ++i) {
            float m = gates[i * 4];
            for (int j = 1; j < 4; ++j) m = fmaxf(m, gates[i * 4 + j]);
            float e[4], sum = 0.f;
            for (int j = 0; j < 4; ++j) { e[j] = expf(gates[i * 4 + j] - m); sum += e[j]; }
            for (int j = 0; j < 4; ++j) cpre += e[j] / sum;
        }
        cpre *= 0.25f;

        // coeff_post: mean_i sum_j softmax_col(post)
        float cpost = 0.f;
        for (int j = 0; j < 4; ++j) {
            float m = gates[16 + j];
            for (int i = 1; i < 4; ++i) m = fmaxf(m, gates[16 + i * 4 + j]);
            float e[4], sum = 0.f;
            for (int i = 0; i < 4; ++i) { e[i] = expf(gates[16 + i * 4 + j] - m); sum += e[i]; }
            for (int i = 0; i < 4; ++i) cpost += e[i] / sum;
        }
        cpost *= 0.25f;

        // coeff_res: iterative Cayley on res (4x4), 2 iterations
        float w[16], y[16];
        for (int i = 0; i < 4; ++i)
            for (int j = 0; j < 4; ++j)
                w[i * 4 + j] = gates[32 + i * 4 + j] - gates[32 + j * 4 + i];
        for (int i = 0; i < 4; ++i)
            for (int j = 0; j < 4; ++j)
                y[i * 4 + j] = (i == j ? 1.f : 0.f) + 0.1f * w[i * 4 + j];
        for (int it = 0; it < 2; ++it) {
            float t[16];
            for (int i = 0; i < 16; ++i) t[i] = y[i];
            for (int i = 0; i < 4; ++i) t[i * 4 + i] += 1.f;   // I + y
            float yn[16];
            for (int i = 0; i < 4; ++i)
                for (int j = 0; j < 4; ++j) {
                    float s2 = 0.f;
                    for (int k = 0; k < 4; ++k) s2 += w[i * 4 + k] * t[k * 4 + j];
                    yn[i * 4 + j] = (i == j ? 1.f : 0.f) + 0.05f * s2;
                }
            for (int i = 0; i < 16; ++i) y[i] = yn[i];
        }
        float cres = 0.f;
        for (int i = 0; i < 16; ++i) cres += y[i];
        cres *= 0.25f;

        c_res[tok] = cres;
        c_mix[tok] = cpre * cpost;
    }
}

// ---------------------------------------------------------------------------
// Kernel 2: out = c_res[m]*x + c_mix[m]*(x @ W_sub)
// bf16 hi/lo split (a = hi + lo): acc += hi*hi + hi*lo + lo*hi  (fp32-class
// accuracy at bf16 WMMA rate; 24 v_wmma per K-step per wave).
// Double-buffered LDS, one barrier per K-step; next tile's global loads stay
// in flight behind the current tile's WMMAs; all LDS stores are b64/b128.
// ---------------------------------------------------------------------------
__global__ void __launch_bounds__(256)
gemm_kernel(const float* __restrict__ x, const float* __restrict__ W,
            const float* __restrict__ c_res, const float* __restrict__ c_mix,
            float* __restrict__ out)
{
    __shared__ __bf16 Ah[2][BM][LDB];   // A tile hi, row-major [m][k]
    __shared__ __bf16 Al[2][BM][LDB];   // A tile lo
    __shared__ __bf16 Bh[2][BN][LDB];   // B tile hi, TRANSPOSED [n][k]
    __shared__ __bf16 Bl[2][BN][LDB];   // B tile lo

    const int tid = threadIdx.x;
    const int n0  = blockIdx.x * BN;
    const int m0  = blockIdx.y * BM;

    const int wave  = tid >> 5;           // wave32
    const int lane  = tid & 31;
    const int wm    = (wave & 3) * 32;    // 4 waves along M -> 32 rows each
    const int wn    = (wave >> 2) * 64;   // 2 waves along N -> 64 cols each
    const int lr    = lane & 15;
    const int khalf = lane >> 4;

    // A loader: thread reads 4x float4 along K (rows ar+i*32, cols ac..ac+3)
    const int ar = tid >> 3;              // base row 0..31 (+i*32)
    const int ac = (tid & 7) * 4;         // col in [0,BK)
    // B loader: thread owns one N column, 16 contiguous K values
    //   global: 16 coalesced b32 loads (32 lanes on consecutive n)
    //   LDS:    2x ds_store_b128 per hi/lo (contiguous [n][k] row segment)
    const int bn2 = tid & 127;            // column n within BN
    const int bkb = (tid >> 7) * 16;      // k base: 0 or 16

    float4 areg[4];                       // staged A (next tile)
    float  breg[16];                      // staged B (next tile)

    auto load_global = [&](int kt) {
        const int k0 = kt * BK;
        #pragma unroll
        for (int i = 0; i < 4; ++i)
            areg[i] = *(const float4*)(x + (size_t)(m0 + ar + i * 32) * K_DIM + k0 + ac);
        #pragma unroll
        for (int i = 0; i < 16; ++i)
            breg[i] = W[(size_t)(k0 + bkb + i) * N_DIM + n0 + bn2];
    };

    auto store_lds = [&](int buf) {
        #pragma unroll
        for (int i = 0; i < 4; ++i) {
            const int row = ar + i * 32;
            const float f[4] = {areg[i].x, areg[i].y, areg[i].z, areg[i].w};
            __bf16 h[4], l[4];
            #pragma unroll
            for (int j = 0; j < 4; ++j) split_bf16(f[j], h[j], l[j]);
            uint2 hq = {pack2(h[0], h[1]), pack2(h[2], h[3])};
            uint2 lq = {pack2(l[0], l[1]), pack2(l[2], l[3])};
            *(uint2*)&Ah[buf][row][ac] = hq;       // ds_store_b64
            *(uint2*)&Al[buf][row][ac] = lq;
        }
        {
            __bf16 h[16], l[16];
            #pragma unroll
            for (int i = 0; i < 16; ++i) split_bf16(breg[i], h[i], l[i]);
            uint4 h0 = {pack2(h[0],h[1]),  pack2(h[2],h[3]),
                        pack2(h[4],h[5]),  pack2(h[6],h[7])};
            uint4 h1 = {pack2(h[8],h[9]),  pack2(h[10],h[11]),
                        pack2(h[12],h[13]),pack2(h[14],h[15])};
            uint4 l0 = {pack2(l[0],l[1]),  pack2(l[2],l[3]),
                        pack2(l[4],l[5]),  pack2(l[6],l[7])};
            uint4 l1 = {pack2(l[8],l[9]),  pack2(l[10],l[11]),
                        pack2(l[12],l[13]),pack2(l[14],l[15])};
            *(uint4*)&Bh[buf][bn2][bkb]     = h0;  // ds_store_b128
            *(uint4*)&Bh[buf][bn2][bkb + 8] = h1;
            *(uint4*)&Bl[buf][bn2][bkb]     = l0;
            *(uint4*)&Bl[buf][bn2][bkb + 8] = l1;
        }
    };

    v8f acc[2][4] = {};

    // Prologue: tile 0 -> buf 0
    load_global(0);
    store_lds(0);

    for (int kt = 0; kt < KTILES; ++kt) {
        const int cur = kt & 1;

        // Issue next tile's global loads; they complete behind the WMMAs.
        if (kt + 1 < KTILES) load_global(kt + 1);

        // Single barrier per iteration:
        //  - publishes buf[cur] stores from the previous iteration
        //  - guarantees all waves finished *reading* buf[1-cur] (their frag
        //    ds_loads preceded this barrier; s_wait_dscnt 0 precedes barrier)
        __syncthreads();

        // A fragments: lane holds row (wm+mt*16+lr), K = {ks..ks+7, ks+16..ks+23}
        Frag ah[2], al[2];
        #pragma unroll
        for (int mt = 0; mt < 2; ++mt) {
            const int row = wm + mt * 16 + lr;
            const int ks  = khalf * 8;
            ah[mt].q[0] = *(const uint4*)&Ah[cur][row][ks];
            ah[mt].q[1] = *(const uint4*)&Ah[cur][row][ks + 16];
            al[mt].q[0] = *(const uint4*)&Al[cur][row][ks];
            al[mt].q[1] = *(const uint4*)&Al[cur][row][ks + 16];
        }
        // B fragments: lane holds col (wn+nt*16+lr), K = {ks..ks+7, ks+16..ks+23}
        #pragma unroll
        for (int nt = 0; nt < 4; ++nt) {
            const int col = wn + nt * 16 + lr;
            const int ks  = khalf * 8;
            Frag bh, bl;
            bh.q[0] = *(const uint4*)&Bh[cur][col][ks];
            bh.q[1] = *(const uint4*)&Bh[cur][col][ks + 16];
            bl.q[0] = *(const uint4*)&Bl[cur][col][ks];
            bl.q[1] = *(const uint4*)&Bl[cur][col][ks + 16];
            #pragma unroll
            for (int mt = 0; mt < 2; ++mt) {
                acc[mt][nt] = wmma_bf16(ah[mt].v, bh.v, acc[mt][nt]);
                acc[mt][nt] = wmma_bf16(ah[mt].v, bl.v, acc[mt][nt]);
                acc[mt][nt] = wmma_bf16(al[mt].v, bh.v, acc[mt][nt]);
            }
        }

        // Convert + store the staged next tile into the other buffer.
        // Safe: buf[1-cur] readers all finished before this iteration's
        // barrier; new contents are only consumed after the next barrier.
        if (kt + 1 < KTILES) store_lds(1 - cur);
    }

    // Epilogue: C/D layout -> row = vgpr + 8*khalf, col = lane&15
    #pragma unroll
    for (int mt = 0; mt < 2; ++mt) {
        #pragma unroll
        for (int nt = 0; nt < 4; ++nt) {
            const int col = n0 + wn + nt * 16 + lr;
            #pragma unroll
            for (int v = 0; v < 8; ++v) {
                const int row = m0 + wm + mt * 16 + khalf * 8 + v;
                const size_t off = (size_t)row * N_DIM + col;
                out[off] = c_res[row] * x[off] + c_mix[row] * acc[mt][nt][v];
            }
        }
    }
}

extern "C" void kernel_launch(void* const* d_in, const int* in_sizes, int n_in,
                              void* d_out, int out_size, void* d_ws, size_t ws_size,
                              hipStream_t stream) {
    (void)in_sizes; (void)n_in; (void)out_size; (void)ws_size;
    const float* x      = (const float*)d_in[0];
    const float* ln_w   = (const float*)d_in[1];
    const float* ln_b   = (const float*)d_in[2];
    const float* proj_w = (const float*)d_in[3];
    const float* proj_b = (const float*)d_in[4];
    const float* W_sub  = (const float*)d_in[5];
    float* out  = (float*)d_out;

    float* cres = (float*)d_ws;          // M_DIM floats
    float* cmix = cres + M_DIM;          // M_DIM floats (64 KB total)

    coeff_kernel<<<M_DIM, 256, 0, stream>>>(x, ln_w, ln_b, proj_w, proj_b, cres, cmix);
    gemm_kernel<<<dim3(N_DIM / BN, M_DIM / BM), 256, 0, stream>>>(x, W_sub, cres, cmix, out);
}
